// SwapPredGNN_76751065579853
// MI455X (gfx1250) — matrix operand order repaired and checked by reading, not compiled, not run on hardware
//
#include <hip/hip_runtime.h>
#include <hip/hip_bf16.h>

typedef __attribute__((ext_vector_type(16))) __bf16 v16bf;
typedef __attribute__((ext_vector_type(8)))  float  v8f;

#define NEG_SLOPE 0.2f

// ---------------- ordered-float <-> uint monotone key (for atomic segment max) ---
__device__ __forceinline__ unsigned f2key(float f) {
  unsigned u = __float_as_uint(f);
  return (u & 0x80000000u) ? ~u : (u | 0x80000000u);
}
__device__ __forceinline__ float key2f(unsigned k) {
  unsigned u = (k & 0x80000000u) ? (k & 0x7fffffffu) : ~k;
  return __uint_as_float(u);
}

// ---------------- pack W (f32 row-major [k][n], 128x128) into bf16 WMMA B-fragment
// order: Wswz[((tile*4 + kk)*32 + lane)*16 + i], where per ISA B striping:
//   n = tile*16 + (lane&15),  k = kk*32 + (i>>3)*16 + (lane>>4)*8 + (i&7)
__global__ void pack_w_kernel(const float* __restrict__ W, __bf16* __restrict__ Wswz) {
  int idx  = blockIdx.x * 256 + threadIdx.x;      // 0 .. 16383
  int i    = idx & 15;
  int lane = (idx >> 4) & 31;
  int kk   = (idx >> 9) & 3;
  int t    = idx >> 11;
  int k = kk * 32 + ((i >> 3) * 16) + ((lane >> 4) * 8) + (i & 7);
  int n = t * 16 + (lane & 15);
  Wswz[idx] = (__bf16)W[k * 128 + n];
}

// ---------------- GEMM h[M,128] = in[M,128] @ W[128,128] via v_wmma_f32_16x16x32_bf16
// block = 256 threads = 8 waves; block computes 32 rows x 128 cols (2 row-tiles).
// Wave w owns cols [16w,16w+16); its 4 B fragments are loaded ONCE into registers
// and reused across both row tiles -> 8 WMMAs per wave, half the L2 W-traffic.
__global__ void __launch_bounds__(256) gemm_kernel(const float* __restrict__ in,
                                                   const __bf16* __restrict__ Wswz,
                                                   float* __restrict__ h) {
  // A fragments swizzled in LDS: sA[rt][(kk*32 + lane)*16 + i]
  __shared__ __attribute__((aligned(32))) __bf16 sA[2][4 * 32 * 16];
  long row0 = (long)blockIdx.x * 32;
  int tid = threadIdx.x;
  const float* src = in + row0 * 128;
#pragma unroll
  for (int j = 0; j < 16; ++j) {
    int idx = tid * 16 + j;            // linear over 32x128 tile (row-major)
    int r = idx >> 7, c = idx & 127;   // r: 0..31
    float f = src[idx];
    // ISA 16-bit A layout: k = (i/8)*16 + half*8 + (i&7), lane = half*16 + row
    int kk = c >> 5, k = c & 31;
    int group = k >> 4, within = k & 15;
    int halfo = within >> 3;
    int ii = group * 8 + (within & 7);
    int lane = halfo * 16 + (r & 15);
    sA[r >> 4][(kk * 32 + lane) * 16 + ii] = (__bf16)f;
  }

  int wave = tid >> 5, lane = tid & 31;
  // load the wave's 4 B fragments once (held in registers across both row tiles)
  v16bf bfr[4];
#pragma unroll
  for (int kk = 0; kk < 4; ++kk)
    bfr[kk] = *(const v16bf*)(Wswz + (((long)wave * 4 + kk) * 32 + lane) * 16);

  __syncthreads();

  int n = lane & 15, mo = (lane >> 4) * 8;
#pragma unroll
  for (int rt = 0; rt < 2; ++rt) {
    v8f acc = {0.f, 0.f, 0.f, 0.f, 0.f, 0.f, 0.f, 0.f};
#pragma unroll
    for (int kk = 0; kk < 4; ++kk) {
      v16bf a = *(const v16bf*)(&sA[rt][(kk * 32 + lane) * 16]);
      acc = __builtin_amdgcn_wmma_f32_16x16x32_bf16(false, a, false, bfr[kk],
                                                    (short)0, acc, false, false);
    }
    // C layout: VGPR r -> row = r + (lane>>4)*8, col = lane&15
    float* dst = h + (row0 + rt * 16 + mo) * 128 + wave * 16 + n;
#pragma unroll
    for (int r = 0; r < 8; ++r) dst[r * 128] = acc[r];
  }
}

// ---------------- per-node attention scalars: one wave per node --------------------
__global__ void alpha_kernel(const float* __restrict__ h, const float* __restrict__ a_s,
                             const float* __restrict__ a_d, float* __restrict__ as_o,
                             float* __restrict__ ad_o, int nnodes) {
  int node = blockIdx.x * 8 + (threadIdx.x >> 5);
  if (node >= nnodes) return;
  int lane = threadIdx.x & 31;
  const float* hp = h + (long)node * 128;
  float s = 0.f, d = 0.f;
#pragma unroll
  for (int j = 0; j < 4; ++j) {
    float v = hp[lane + 32 * j];
    s += v * a_s[lane + 32 * j];
    d += v * a_d[lane + 32 * j];
  }
#pragma unroll
  for (int off = 16; off > 0; off >>= 1) {
    s += __shfl_down(s, off, 32);
    d += __shfl_down(d, off, 32);
  }
  if (lane == 0) { as_o[node] = s; ad_o[node] = d; }
}

// ---------------- init helpers ----------------------------------------------------
__global__ void zero_f32_kernel(float* __restrict__ p, long n) {
  long i = (long)blockIdx.x * blockDim.x + threadIdx.x;
  long stride = (long)gridDim.x * blockDim.x;
  for (; i < n; i += stride) p[i] = 0.f;
}
__global__ void init_md_kernel(unsigned* __restrict__ mkey, float* __restrict__ denom, int n) {
  int i = blockIdx.x * 256 + threadIdx.x;
  if (i < n) { mkey[i] = 0u; denom[i] = 0.f; }
}

// ---------------- edge pass 1: e = leakyrelu(as[src]+ad[dst]); segment max ---------
__global__ void edge_max_kernel(const long long* __restrict__ ei, const float* __restrict__ as,
                                const float* __restrict__ ad, float* __restrict__ e_buf,
                                unsigned* __restrict__ mkey, int E_, int Etot_) {
  int e = blockIdx.x * 256 + threadIdx.x;
  if (e >= Etot_) return;
  int s, d;
  if (e < E_) { s = (int)ei[e]; d = (int)ei[E_ + e]; }
  else        { s = e - E_; d = s; }
  float v = as[s] + ad[d];
  v = (v > 0.f) ? v : NEG_SLOPE * v;
  e_buf[e] = v;
  atomicMax(mkey + d, f2key(v));
}

// ---------------- edge pass 2: ex = exp(e - m[dst]); segment sum -------------------
__global__ void edge_exp_kernel(const long long* __restrict__ ei, const unsigned* __restrict__ mkey,
                                float* __restrict__ e_buf, float* __restrict__ denom,
                                int E_, int Etot_) {
  int e = blockIdx.x * 256 + threadIdx.x;
  if (e >= Etot_) return;
  int d = (e < E_) ? (int)ei[E_ + e] : (e - E_);
  float ex = __expf(e_buf[e] - key2f(mkey[d]));
  e_buf[e] = ex;
  atomicAdd(denom + d, ex);
}

// ---------------- edge pass 3: acc[dst] += h[src] * (ex / denom[dst]) --------------
// one wave per edge; 4 coalesced f32 atomics per lane (128 cols / 32 lanes)
__global__ void scatter_kernel(const long long* __restrict__ ei, const float* __restrict__ e_buf,
                               const float* __restrict__ denom, const float* __restrict__ h,
                               float* __restrict__ acc, int E_, int Etot_) {
  int edge = blockIdx.x * 8 + (threadIdx.x >> 5);
  if (edge >= Etot_) return;
  int lane = threadIdx.x & 31;
  int s, d;
  if (edge < E_) { s = (int)ei[edge]; d = (int)ei[E_ + edge]; }
  else           { s = d = edge - E_; }
  float w = e_buf[edge] / denom[d];
  const float* hs = h + (long)s * 128;
  float* od = acc + (long)d * 128;
#pragma unroll
  for (int j = 0; j < 4; ++j)
    atomicAdd(od + lane + 32 * j, hs[lane + 32 * j] * w);
}

// ---------------- bias + optional relu --------------------------------------------
__global__ void finalize_kernel(float* __restrict__ acc, const float* __restrict__ b,
                                long n128, int relu) {
  long i = (long)blockIdx.x * 256 + threadIdx.x;
  if (i >= n128) return;
  float v = acc[i] + b[(int)(i & 127)];
  if (relu) v = fmaxf(v, 0.f);
  acc[i] = v;
}

extern "C" void kernel_launch(void* const* d_in, const int* in_sizes, int n_in,
                              void* d_out, int out_size, void* d_ws, size_t ws_size,
                              hipStream_t stream) {
  const float*      x  = (const float*)d_in[0];
  const long long*  ei = (const long long*)d_in[1];   // int64 [2, E]
  const float* W[3]  = {(const float*)d_in[2], (const float*)d_in[6],  (const float*)d_in[10]};
  const float* As[3] = {(const float*)d_in[3], (const float*)d_in[7],  (const float*)d_in[11]};
  const float* Ad[3] = {(const float*)d_in[4], (const float*)d_in[8],  (const float*)d_in[12]};
  const float* Bb[3] = {(const float*)d_in[5], (const float*)d_in[9],  (const float*)d_in[13]};

  const int  NN = in_sizes[0] / 128;   // 100000
  const int  EE = in_sizes[1] / 2;     // 1600000
  const int  ET = EE + NN;             // + self loops
  const long NF = (long)NN * 128;

  // scratch carve-out (all 256B-aligned): ~111 MB
  char* ws = (char*)d_ws;
  size_t off = 0;
  auto carve = [&](size_t bytes) -> void* {
    off = (off + 255) & ~(size_t)255;
    void* p = ws + off;
    off += bytes;
    return p;
  };
  float*    bufH  = (float*)carve(NF * sizeof(float));       // h = in @ W
  float*    bufA  = (float*)carve(NF * sizeof(float));       // layer output accumulator
  float*    as_o  = (float*)carve((size_t)NN * sizeof(float));
  float*    ad_o  = (float*)carve((size_t)NN * sizeof(float));
  unsigned* mkey  = (unsigned*)carve((size_t)NN * sizeof(unsigned));
  float*    denom = (float*)carve((size_t)NN * sizeof(float));
  float*    e_buf = (float*)carve((size_t)ET * sizeof(float));
  __bf16*   Wswz  = (__bf16*)carve((size_t)128 * 128 * sizeof(__bf16));
  (void)ws_size; (void)n_in; (void)out_size;

  const float* in_feat = x;
  for (int L = 0; L < 3; ++L) {
    float* acc = (L == 2) ? (float*)d_out : bufA;

    pack_w_kernel<<<64, 256, 0, stream>>>(W[L], Wswz);
    gemm_kernel<<<NN / 32, 256, 0, stream>>>(in_feat, Wswz, bufH);
    alpha_kernel<<<(NN + 7) / 8, 256, 0, stream>>>(bufH, As[L], Ad[L], as_o, ad_o, NN);
    zero_f32_kernel<<<2048, 256, 0, stream>>>(acc, NF);          // also frees bufA for reuse
    init_md_kernel<<<(NN + 255) / 256, 256, 0, stream>>>(mkey, denom, NN);
    edge_max_kernel<<<(ET + 255) / 256, 256, 0, stream>>>(ei, as_o, ad_o, e_buf, mkey, EE, ET);
    edge_exp_kernel<<<(ET + 255) / 256, 256, 0, stream>>>(ei, mkey, e_buf, denom, EE, ET);
    scatter_kernel<<<(ET + 7) / 8, 256, 0, stream>>>(ei, e_buf, denom, bufH, acc, EE, ET);
    finalize_kernel<<<(NF + 255) / 256, 256, 0, stream>>>(acc, Bb[L], NF, (L < 2) ? 1 : 0);

    in_feat = acc;   // for L=0,1 this is bufA; layer-2 writes d_out
  }
}
